// seq2seq_19043884990776
// MI455X (gfx1250) — compile-verified, matrix operand order checked by reading, not run on hardware
//
#include <hip/hip_runtime.h>

// ---------------------------------------------------------------------------
// seq2seq LSTM forward for gfx1250 (MI455X), wave32 + WMMA bf16 path.
//   * gates accumulate f32 via v_wmma_f32_16x16x32_bf16, A/B operands bf16
//   * input-side gate GEMMs batched over all 48 timesteps (parallel)
//   * recurrent scans sharded over G workgroups (G=16 dec, 8/dir enc):
//     each block owns a hidden-unit slice, K split 4-ways across its waves,
//     h exchanged per step through a double-buffered global bf16 buffer with
//     an agent-scope release/acquire atomic barrier (single-WGP scans would
//     be limited by one WGP's L2 fetch bandwidth on the 2 MB/step weights)
//   * recurrent weights pre-converted to bf16 once per launch
// ---------------------------------------------------------------------------

typedef __bf16 bf16_t;
typedef __attribute__((ext_vector_type(16))) __bf16 v16bf;
typedef __attribute__((ext_vector_type(8)))  __bf16 v8bf;
typedef __attribute__((ext_vector_type(8)))  float  v8f;
typedef __attribute__((ext_vector_type(16))) float  v16f;

#define NB   16      // batch
#define TS   48      // seq len (src == tgt)
#define HH   512     // hidden
#define HDE  256     // encoder per-direction hidden
#define NL   16      // layers
#define NV   32000   // vocab
#define GENC 8       // blocks per encoder direction
#define GDEC 16      // blocks for decoder scan

__device__ __forceinline__ float sigf(float x) { return 1.0f / (1.0f + __expf(-x)); }

// load 16 consecutive f32 and convert to a bf16 WMMA fragment slice
__device__ __forceinline__ v16bf cvt16(const float* __restrict__ p) {
  v16f f = *(const v16f*)p;
  v16bf r;
#pragma unroll
  for (int i = 0; i < 16; ++i) r[i] = (__bf16)f[i];
  return r;
}

// inter-workgroup rendezvous: release our h-slice stores, acquire others'.
__device__ __forceinline__ void gbarrier(unsigned* cnt, unsigned target) {
  __syncthreads();
  if (threadIdx.x == 0) {
    __hip_atomic_fetch_add(cnt, 1u, __ATOMIC_ACQ_REL, __HIP_MEMORY_SCOPE_AGENT);
    while (__hip_atomic_load(cnt, __ATOMIC_ACQUIRE, __HIP_MEMORY_SCOPE_AGENT) < target)
      __builtin_amdgcn_s_sleep(2);
  }
  __syncthreads();
}

// ---------------------------------------------------------------------------
// Bulk f32 -> bf16 conversion (8 elems / thread).  n8 = n/8.
// ---------------------------------------------------------------------------
__global__ void convert_bf16_kernel(const float* __restrict__ src,
                                    bf16_t* __restrict__ dst, int n8) {
  int i = blockIdx.x * blockDim.x + threadIdx.x;
  if (i >= n8) return;
  v8f f = *(const v8f*)(src + (size_t)i * 8);
  v8bf o;
#pragma unroll
  for (int j = 0; j < 8; ++j) o[j] = (__bf16)f[j];
  *(v8bf*)(dst + (size_t)i * 8) = o;
}

// ---------------------------------------------------------------------------
// Embedding gather: out (T, B, H) bf16 from ids (B, T) int32.
// ---------------------------------------------------------------------------
__global__ void embed_kernel(const int* __restrict__ ids,
                             const float* __restrict__ emb,
                             bf16_t* __restrict__ outv) {
  int e = blockIdx.x * blockDim.x + threadIdx.x;
  const int total = TS * NB * HH;
  if (e >= total) return;
  int hcol = e & (HH - 1);
  int m = e >> 9;            // H == 512
  int t = m >> 4, bb = m & 15;
  int id = ids[bb * TS + t];
  outv[e] = (__bf16)emb[(size_t)id * HH + hcol];
}

// ---------------------------------------------------------------------------
// G(M,N) = A(M,K)bf16 @ W(N,K)^T f32 (+ bias).  One wave per 16x16 tile.
// W converted f32->bf16 in the load path (one-pass parallel GEMM).
// ---------------------------------------------------------------------------
__global__ __launch_bounds__(256, 1)
void gemm_gates_kernel(const bf16_t* __restrict__ A, const float* __restrict__ W,
                       const float* __restrict__ bias, float* __restrict__ G,
                       int M, int N, int K) {
  const int lane = threadIdx.x & 31;
  const int wid  = blockIdx.x * (blockDim.x >> 5) + (threadIdx.x >> 5);
  const int tilesN = N >> 4;
  const int totalTiles = (M >> 4) * tilesN;
  if (wid >= totalTiles) return;                 // uniform per wave
  const int tm = wid / tilesN, tn = wid % tilesN;
  const int m0 = tm << 4, n0 = tn << 4;
  const int lrow = lane & 15, lhi = lane >> 4;
  const int col  = n0 + lrow;

  v8f acc;
  const float bv = bias ? bias[col] : 0.0f;
#pragma unroll
  for (int r = 0; r < 8; ++r) acc[r] = bv;

  const bf16_t* Ap = A + (size_t)(m0 + lrow) * K + lhi * 16;
  const float*  Wp = W + (size_t)col * K + lhi * 16;
  for (int k0 = 0; k0 < K; k0 += 32) {
    __builtin_prefetch(Wp + 256, 0, 1);          // global_prefetch_b8
    v16bf a = *(const v16bf*)Ap;
    v16bf b = cvt16(Wp);
    acc = __builtin_amdgcn_wmma_f32_16x16x32_bf16(false, a, false, b,
                                                  (short)0, acc, false, false);
    Ap += 32; Wp += 32;
  }
#pragma unroll
  for (int r = 0; r < 8; ++r)
    G[(size_t)(m0 + r + 8 * lhi) * N + col] = acc[r];
}

// ---------------------------------------------------------------------------
// Logits: out[b, v, t] = ds(M,K)bf16 @ lin_W(V,K)^T + lin_b.  m = t*16 + b.
// ---------------------------------------------------------------------------
__global__ __launch_bounds__(256, 1)
void gemm_logits_kernel(const bf16_t* __restrict__ A, const float* __restrict__ W,
                        const float* __restrict__ bias, float* __restrict__ out) {
  const int M = TS * NB, N = NV, K = HH;
  const int lane = threadIdx.x & 31;
  const int wid  = blockIdx.x * (blockDim.x >> 5) + (threadIdx.x >> 5);
  const int tilesN = N >> 4;
  const int totalTiles = (M >> 4) * tilesN;
  if (wid >= totalTiles) return;
  const int tm = wid / tilesN, tn = wid % tilesN;
  const int m0 = tm << 4, n0 = tn << 4;
  const int lrow = lane & 15, lhi = lane >> 4;
  const int col  = n0 + lrow;

  v8f acc;
  const float bv = bias[col];
#pragma unroll
  for (int r = 0; r < 8; ++r) acc[r] = bv;

  const bf16_t* Ap = A + (size_t)(m0 + lrow) * K + lhi * 16;
  const float*  Wp = W + (size_t)col * K + lhi * 16;
  for (int k0 = 0; k0 < K; k0 += 32) {
    __builtin_prefetch(Wp + 256, 0, 1);
    v16bf a = *(const v16bf*)Ap;
    v16bf b = cvt16(Wp);
    acc = __builtin_amdgcn_wmma_f32_16x16x32_bf16(false, a, false, b,
                                                  (short)0, acc, false, false);
    Ap += 32; Wp += 32;
  }
#pragma unroll
  for (int r = 0; r < 8; ++r) {
    const int m = m0 + r + 8 * lhi;
    const int t = m >> 4, bb = m & 15;
    out[(size_t)bb * NV * TS + (size_t)col * TS + t] = acc[r];
  }
}

// ---------------------------------------------------------------------------
// Encoder recurrent scan, sharded: grid = 2*GENC blocks (dir = blk/GENC).
// Block owns JS=32 hidden units -> 8 gate tiles; 32 waves = 8 tiles x 4 K-
// quarters into LDS partial accumulators.  h exchanged via hglob (per dir,
// double-buffered) + atomic barrier.
// ---------------------------------------------------------------------------
__global__ __launch_bounds__(1024, 1)
void enc_scan_kernel(const float* __restrict__ gih_f,    // (T,16,1024)
                     const float* __restrict__ gih_b,    // (T,16,1024)
                     const bf16_t* __restrict__ Whh,     // (2,1024,256) bf16
                     bf16_t* __restrict__ xs_out,        // (T,16,512)
                     float* __restrict__ h_out,          // (16,512) layer slot
                     float* __restrict__ c_out,          // (16,512)
                     bf16_t* __restrict__ hglob,         // (2dir,2phase,16,256)
                     unsigned* __restrict__ barcnt) {    // 2 slots (per dir)
  const int K = HDE, NG = 4 * HDE, JS = HDE / GENC /*32*/, LC = 4 * JS /*128*/;
  const int dir = blockIdx.x / GENC;
  const int g   = blockIdx.x % GENC;
  const int j0  = g * JS;
  const float*  gih = dir ? gih_b : gih_f;
  const bf16_t* W   = Whh + (size_t)dir * NG * K;
  bf16_t*   hg  = hglob + (size_t)dir * 2 * 16 * K;
  unsigned* cnt = barcnt + dir;

  __shared__ float  gp[4 * 16 * 128];        // 4 K-quarters x 16 rows x LC (32 KB)
  __shared__ bf16_t hL[16 * HDE];            // full h, bf16 (8 KB)

  const int tid  = threadIdx.x;
  const int wave = tid >> 5, lane = tid & 31;
  const int lrow = lane & 15, lhi = lane >> 4;
  // wave -> (tile, K-quarter): 8 tiles x 4 quarters
  const int tt = wave >> 2, kq = wave & 3;
  const int q  = tt >> 1, jt = tt & 1;       // gate quadrant, tile-in-quadrant
  const int col  = q * HDE + j0 + jt * 16 + lrow;   // global gate column
  const int lcol = q * JS  + jt * 16 + lrow;        // local column (0..127)
  // per-thread owned (b, j): 16*JS = 512 elements
  const bool owned = tid < 16 * JS;
  const int obb = tid >> 5, ojj = tid & (JS - 1);
  const int oj  = j0 + ojj;

  float creg = 0.0f, hlast = 0.0f;
  if (owned) hg[obb * K + oj] = (__bf16)0.0f;       // phase-0 h = 0
  unsigned tgt = GENC;
  gbarrier(cnt, tgt); tgt += GENC;

  int phase = 0;
  for (int s = 0; s < TS; ++s) {
    // pull full h (all blocks' slices) into LDS
    {
      const unsigned* src = (const unsigned*)(hg + phase * 16 * K);
      unsigned* dst = (unsigned*)hL;
      for (int i = tid; i < 16 * K / 2; i += 1024) dst[i] = src[i];
    }
    __syncthreads();
    const int t_in = dir ? (TS - 1 - s) : s;
    // partial gates: this wave's tile, K-quarter kq (K/4 = 64 -> 2 k-steps)
    {
      v8f acc;
      if (kq == 0) {
#pragma unroll
        for (int r = 0; r < 8; ++r)
          acc[r] = gih[(size_t)(t_in * 16 + r + 8 * lhi) * NG + col];
      } else {
#pragma unroll
        for (int r = 0; r < 8; ++r) acc[r] = 0.0f;
      }
      const bf16_t* hp = hL + lrow * K + kq * (K / 4) + lhi * 16;
      const bf16_t* Wp = W + (size_t)col * K + kq * (K / 4) + lhi * 16;
#pragma unroll
      for (int k0 = 0; k0 < K / 4; k0 += 32) {
        v16bf a = *(const v16bf*)(hp + k0);
        v16bf b = *(const v16bf*)(Wp + k0);
        acc = __builtin_amdgcn_wmma_f32_16x16x32_bf16(false, a, false, b,
                                                      (short)0, acc, false, false);
      }
#pragma unroll
      for (int r = 0; r < 8; ++r)
        gp[(kq * 16 + r + 8 * lhi) * LC + lcol] = acc[r];
    }
    __syncthreads();
    // elementwise LSTM cell on owned slice
    if (owned) {
      float gi = 0.f, gf = 0.f, gg = 0.f, go = 0.f;
#pragma unroll
      for (int kk = 0; kk < 4; ++kk) {
        gi += gp[(kk * 16 + obb) * LC + 0 * JS + ojj];
        gf += gp[(kk * 16 + obb) * LC + 1 * JS + ojj];
        gg += gp[(kk * 16 + obb) * LC + 2 * JS + ojj];
        go += gp[(kk * 16 + obb) * LC + 3 * JS + ojj];
      }
      const float i_ = sigf(gi), f_ = sigf(gf), o_ = sigf(go);
      creg = f_ * creg + i_ * tanhf(gg);
      const float hh = o_ * tanhf(creg);
      hlast = hh;
      hg[(phase ^ 1) * 16 * K + obb * K + oj] = (__bf16)hh;
      const int t_out = dir ? (TS - 1 - s) : s;
      xs_out[(size_t)(t_out * 16 + obb) * HH + dir * HDE + oj] = (__bf16)hh;
    }
    gbarrier(cnt, tgt); tgt += GENC;
    phase ^= 1;
  }
  // final (hT, cT) -> decoder init state, concat fwd/bwd halves
  if (owned) {
    h_out[obb * HH + dir * HDE + oj] = hlast;
    c_out[obb * HH + dir * HDE + oj] = creg;
  }
}

// ---------------------------------------------------------------------------
// Decoder recurrent scan, sharded over GDEC=16 blocks. JS=32 -> 8 gate tiles;
// 32 waves = 8 tiles x 4 K-quarters (K/4 = 128 -> 4 k-steps each).
// ---------------------------------------------------------------------------
__global__ __launch_bounds__(1024, 1)
void dec_scan_kernel(const float* __restrict__ gih,    // (T,16,2048)
                     const bf16_t* __restrict__ Whh,   // (2048,512) bf16
                     const float* __restrict__ h0,     // (16,512)
                     const float* __restrict__ c0,     // (16,512)
                     bf16_t* __restrict__ ds_out,      // (T,16,512)
                     bf16_t* __restrict__ hglob,       // (2phase,16,512)
                     unsigned* __restrict__ barcnt) {  // 1 slot
  const int K = HH, NG = 4 * HH, JS = HH / GDEC /*32*/, LC = 4 * JS /*128*/;
  const int g  = blockIdx.x;
  const int j0 = g * JS;

  __shared__ float  gp[4 * 16 * 128];        // 32 KB
  __shared__ bf16_t hL[16 * HH];             // 16 KB

  const int tid  = threadIdx.x;
  const int wave = tid >> 5, lane = tid & 31;
  const int lrow = lane & 15, lhi = lane >> 4;
  const int tt = wave >> 2, kq = wave & 3;
  const int q  = tt >> 1, jt = tt & 1;
  const int col  = q * HH + j0 + jt * 16 + lrow;
  const int lcol = q * JS + jt * 16 + lrow;
  const bool owned = tid < 16 * JS;
  const int obb = tid >> 5, ojj = tid & (JS - 1);
  const int oj  = j0 + ojj;

  float creg = 0.0f;
  if (owned) {
    creg = c0[obb * HH + oj];
    hglob[obb * HH + oj] = (__bf16)h0[obb * HH + oj];   // phase-0 h
  }
  unsigned tgt = GDEC;
  gbarrier(barcnt, tgt); tgt += GDEC;

  int phase = 0;
  for (int s = 0; s < TS; ++s) {
    {
      const unsigned* src = (const unsigned*)(hglob + phase * 16 * K);
      unsigned* dst = (unsigned*)hL;
      for (int i = tid; i < 16 * K / 2; i += 1024) dst[i] = src[i];
    }
    __syncthreads();
    {
      v8f acc;
      if (kq == 0) {
#pragma unroll
        for (int r = 0; r < 8; ++r)
          acc[r] = gih[(size_t)(s * 16 + r + 8 * lhi) * NG + col];
      } else {
#pragma unroll
        for (int r = 0; r < 8; ++r) acc[r] = 0.0f;
      }
      const bf16_t* hp = hL + lrow * K + kq * (K / 4) + lhi * 16;
      const bf16_t* Wp = Whh + (size_t)col * K + kq * (K / 4) + lhi * 16;
#pragma unroll
      for (int k0 = 0; k0 < K / 4; k0 += 32) {
        v16bf a = *(const v16bf*)(hp + k0);
        v16bf b = *(const v16bf*)(Wp + k0);
        acc = __builtin_amdgcn_wmma_f32_16x16x32_bf16(false, a, false, b,
                                                      (short)0, acc, false, false);
      }
#pragma unroll
      for (int r = 0; r < 8; ++r)
        gp[(kq * 16 + r + 8 * lhi) * LC + lcol] = acc[r];
    }
    __syncthreads();
    if (owned) {
      float gi = 0.f, gf = 0.f, gg = 0.f, go = 0.f;
#pragma unroll
      for (int kk = 0; kk < 4; ++kk) {
        gi += gp[(kk * 16 + obb) * LC + 0 * JS + ojj];
        gf += gp[(kk * 16 + obb) * LC + 1 * JS + ojj];
        gg += gp[(kk * 16 + obb) * LC + 2 * JS + ojj];
        go += gp[(kk * 16 + obb) * LC + 3 * JS + ojj];
      }
      const float i_ = sigf(gi), f_ = sigf(gf), o_ = sigf(go);
      creg = f_ * creg + i_ * tanhf(gg);
      const float hh = o_ * tanhf(creg);
      hglob[(phase ^ 1) * 16 * K + obb * K + oj] = (__bf16)hh;
      ds_out[(size_t)(s * 16 + obb) * HH + oj] = (__bf16)hh;
    }
    gbarrier(barcnt, tgt); tgt += GDEC;
    phase ^= 1;
  }
}

// ---------------------------------------------------------------------------
// Host orchestration
// ---------------------------------------------------------------------------
extern "C" void kernel_launch(void* const* d_in, const int* in_sizes, int n_in,
                              void* d_out, int out_size, void* d_ws, size_t ws_size,
                              hipStream_t stream) {
  const int*   x        = (const int*)d_in[0];
  const int*   y        = (const int*)d_in[1];
  const float* enc_emb  = (const float*)d_in[2];
  const float* enc_W_ih = (const float*)d_in[3];   // (L,2,1024,512)
  const float* enc_W_hh = (const float*)d_in[4];   // (L,2,1024,256)
  const float* enc_b    = (const float*)d_in[5];   // (L,2,1024)
  const float* dec_emb  = (const float*)d_in[6];
  const float* dec_W_ih = (const float*)d_in[7];   // (L,2048,512)
  const float* dec_W_hh = (const float*)d_in[8];   // (L,2048,512)
  const float* dec_b    = (const float*)d_in[9];   // (L,2048)
  const float* lin_W    = (const float*)d_in[10];  // (V,512)
  const float* lin_b    = (const float*)d_in[11];  // (V)
  float* out = (float*)d_out;

  // workspace carve-out (256B aligned)
  char* ws = (char*)d_ws;
  size_t off = 0;
  auto carve = [&](size_t bytes) -> void* {
    void* p = ws + off;
    off += (bytes + 255) & ~(size_t)255;
    return p;
  };
  const size_t actElems    = (size_t)TS * NB * HH;              //   393,216
  const size_t encWhhElems = (size_t)NL * 2 * 1024 * HDE;       // 8,388,608
  const size_t decWhhElems = (size_t)NL * 2048 * HH;            // 16,777,216
  bf16_t* xsA    = (bf16_t*)carve(actElems * sizeof(bf16_t));
  bf16_t* xsB    = (bf16_t*)carve(actElems * sizeof(bf16_t));
  bf16_t* dsA    = (bf16_t*)carve(actElems * sizeof(bf16_t));
  bf16_t* dsB    = (bf16_t*)carve(actElems * sizeof(bf16_t));
  float*  gF     = (float*)carve((size_t)TS * NB * 4 * HDE * sizeof(float));
  float*  gBw    = (float*)carve((size_t)TS * NB * 4 * HDE * sizeof(float));
  float*  gD     = (float*)carve((size_t)TS * NB * 4 * HH  * sizeof(float));
  float*  h0s    = (float*)carve((size_t)NL * NB * HH * sizeof(float));
  float*  c0s    = (float*)carve((size_t)NL * NB * HH * sizeof(float));
  bf16_t* whhEnc = (bf16_t*)carve(encWhhElems * sizeof(bf16_t));
  bf16_t* whhDec = (bf16_t*)carve(decWhhElems * sizeof(bf16_t));
  bf16_t* hglE   = (bf16_t*)carve((size_t)2 * 2 * NB * HDE * sizeof(bf16_t));
  bf16_t* hglD   = (bf16_t*)carve((size_t)2 * NB * HH * sizeof(bf16_t));
  unsigned* bars = (unsigned*)carve((size_t)(2 * NL + NL) * sizeof(unsigned));
  (void)ws_size; (void)in_sizes; (void)n_in; (void)out_size;

  // reset all barrier counters once (capture-safe stream op)
  hipMemsetAsync(bars, 0, (2 * NL + NL) * sizeof(unsigned), stream);

  // ---- one-shot recurrent-weight conversion to bf16 ----
  convert_bf16_kernel<<<(int)(encWhhElems / 8 + 255) / 256, 256, 0, stream>>>(
      enc_W_hh, whhEnc, (int)(encWhhElems / 8));
  convert_bf16_kernel<<<(int)(decWhhElems / 8 + 255) / 256, 256, 0, stream>>>(
      dec_W_hh, whhDec, (int)(decWhhElems / 8));

  // ---- encoder ----
  embed_kernel<<<(TS * NB * HH + 255) / 256, 256, 0, stream>>>(x, enc_emb, xsA);
  bf16_t* xsIn = xsA; bf16_t* xsOut = xsB;
  for (int l = 0; l < NL; ++l) {
    const float*  wih0 = enc_W_ih + (size_t)l * 2 * 1024 * HH;
    const float*  wih1 = wih0 + (size_t)1024 * HH;
    const float*  bi0  = enc_b + (size_t)l * 2 * 1024;
    const float*  bi1  = bi0 + 1024;
    const bf16_t* whh  = whhEnc + (size_t)l * 2 * 1024 * HDE;
    // input-side gate GEMMs over all timesteps: M=768, N=1024, K=512
    gemm_gates_kernel<<<384, 256, 0, stream>>>(xsIn, wih0, bi0, gF,  TS * NB, 1024, HH);
    gemm_gates_kernel<<<384, 256, 0, stream>>>(xsIn, wih1, bi1, gBw, TS * NB, 1024, HH);
    // sharded recurrent scan, both directions (2*GENC blocks)
    enc_scan_kernel<<<2 * GENC, 1024, 0, stream>>>(
        gF, gBw, whh, xsOut, h0s + (size_t)l * NB * HH, c0s + (size_t)l * NB * HH,
        hglE, bars + 2 * l);
    bf16_t* tmp = xsIn; xsIn = xsOut; xsOut = tmp;
  }

  // ---- decoder ----
  embed_kernel<<<(TS * NB * HH + 255) / 256, 256, 0, stream>>>(y, dec_emb, dsA);
  bf16_t* dsIn = dsA; bf16_t* dsOut = dsB;
  for (int l = 0; l < NL; ++l) {
    const float*  wih = dec_W_ih + (size_t)l * 2048 * HH;
    const bf16_t* whh = whhDec + (size_t)l * 2048 * HH;
    const float*  bi  = dec_b + (size_t)l * 2048;
    gemm_gates_kernel<<<768, 256, 0, stream>>>(dsIn, wih, bi, gD, TS * NB, 2048, HH);
    dec_scan_kernel<<<GDEC, 1024, 0, stream>>>(
        gD, whh, h0s + (size_t)l * NB * HH, c0s + (size_t)l * NB * HH, dsOut,
        hglD, bars + 2 * NL + l);
    bf16_t* tmp = dsIn; dsIn = dsOut; dsOut = tmp;
  }

  // ---- output projection: (768 x 32000 x 512), store (B, V, T) ----
  gemm_logits_kernel<<<12000, 256, 0, stream>>>(dsIn, lin_W, lin_b, out);
}
